// SaliencyNetworkSPADE_62113817034947
// MI455X (gfx1250) — matrix operand order repaired and checked by reading, not compiled, not run on hardware
//
#include <hip/hip_runtime.h>

typedef __attribute__((ext_vector_type(16))) _Float16 v16h;
typedef __attribute__((ext_vector_type(8)))  float    v8f;

#define NB    8
#define PIX   1024          // 32*32
#define NTOT  8192          // NB*PIX
#define NTIL  512           // NTOT/16 column tiles
#define NSEGC 32
#define HIDC  128
#define CIN   2048

// ---- B-matrix fragment swizzle: element (k, n) of a [K][8192] matrix ----
// lives at ((kt*512 + nt)*32 + lane)*16 + e with kt=k/32, kk=k%32,
// lane = (kk&16)|(n&15), e = kk&15, nt = n/16  (contiguous k order, matches A packing)
__device__ __forceinline__ size_t bswz(int k, int n) {
  int kt = k >> 5, kk = k & 31;
  return ((((size_t)kt << 9) + (n >> 4)) * 32 + ((kk & 16) | (n & 15))) * 16 + (kk & 15);
}

// ---------------------------------------------------------------- packing
// f32 W[M][K] row-major -> f16 A-fragments, M zero-padded to multiple of 16.
// A element (m,k): lane=(m&15)+(khi?16:0), contiguous k order (matches B):
// k = kt*32 + (lane>=16?16:0) + e
__global__ void k_pack_A(const float* __restrict__ W, _Float16* __restrict__ out,
                         int M, int K) {
  const int Ktiles = K >> 5;
  const int total = ((M + 15) >> 4) * Ktiles * 512;
  int idx = blockIdx.x * 256 + threadIdx.x;
  if (idx >= total) return;
  int e = idx & 15, L = (idx >> 4) & 31;
  int tile = idx >> 9;
  int kt = tile % Ktiles, mt = tile / Ktiles;
  int m = mt * 16 + (L & 15);
  int k = kt * 32 + (L & 16) + e;
  out[idx] = (m < M) ? (_Float16)W[(size_t)m * K + k] : (_Float16)0.0f;
}

// one-hot(clip(seg)) -> nearest-resize 32x32 -> 3x3 im2col, B-swizzled.
__global__ void k_segcol(const int* __restrict__ seg, _Float16* __restrict__ out) {
  int idx = blockIdx.x * 256 + threadIdx.x;
  if (idx >= 9 * NSEGC * NTOT) return;
  int n = idx % NTOT, k = idx / NTOT;
  int c = k / 9, r = k % 9;
  int dy = r / 3 - 1, dx = r % 3 - 1;
  int b = n >> 10, p = n & 1023;
  int y = p >> 5, x = p & 31;
  int yy = y + dy, xx = x + dx;
  float v = 0.f;
  if (yy >= 0 && yy < 32 && xx >= 0 && xx < 32) {
    int s = seg[(b * 128 + (yy << 2)) * 128 + (xx << 2)];  // nearest: idx*4
    s = s < 0 ? 0 : (s > NSEGC - 1 ? NSEGC - 1 : s);
    v = (s == c) ? 1.f : 0.f;
  }
  out[bswz(k, n)] = (_Float16)v;
}

// 3x3 im2col of f32 S[128][NTOT] -> f16 B-swizzled [1152][NTOT]
__global__ void k_im2col(const float* __restrict__ S, _Float16* __restrict__ D) {
  int idx = blockIdx.x * 256 + threadIdx.x;
  if (idx >= 9 * HIDC * NTOT) return;
  int n = idx % NTOT, k = idx / NTOT;
  int c = k / 9, r = k % 9;
  int dy = r / 3 - 1, dx = r % 3 - 1;
  int b = n >> 10, p = n & 1023;
  int y = p >> 5, x = p & 31;
  int yy = y + dy, xx = x + dx;
  float v = 0.f;
  if (yy >= 0 && yy < 32 && xx >= 0 && xx < 32)
    v = S[c * NTOT + (b << 10) + (yy << 5) + xx];
  D[bswz(k, n)] = (_Float16)v;
}

// ---------------------------------------------------- WMMA GEMM (wave32)
// C[M,8192] = act( A16[M,K] * B16[K,8192] + bias[M] ). One wave: 16x64 tile,
// 4 v_wmma per K-step, all fragments loaded as contiguous 32B vector loads.
__global__ void __launch_bounds__(32)
k_gemm_wmma(const v16h* __restrict__ A, const v16h* __restrict__ B,
            const float* __restrict__ bias, int M, int Ktiles,
            float* __restrict__ C32, _Float16* __restrict__ C16, int act) {
  const int lane = threadIdx.x;
  const int hi = lane >> 4, lo = lane & 15;
  const int nt0 = blockIdx.x * 4;     // four 16-column tiles
  const int mt = blockIdx.y;

  const v16h* Ap = A + (size_t)mt * Ktiles * 32 + lane;
  const v16h* Bp = B + (size_t)nt0 * 32 + lane;

  v8f acc0 = {}, acc1 = {}, acc2 = {}, acc3 = {};
  for (int kt = 0; kt < Ktiles; ++kt) {
    v16h a  = Ap[(size_t)kt * 32];
    const v16h* Bk = Bp + (size_t)kt * (NTIL * 32);
    v16h b0 = Bk[0];
    v16h b1 = Bk[32];
    v16h b2 = Bk[64];
    v16h b3 = Bk[96];
    __builtin_prefetch(Bk + NTIL * 32, 0, 0);        // next K-step B tiles
    acc0 = __builtin_amdgcn_wmma_f32_16x16x32_f16(false, a, false, b0,
                                                  (short)0, acc0, false, false);
    acc1 = __builtin_amdgcn_wmma_f32_16x16x32_f16(false, a, false, b1,
                                                  (short)0, acc1, false, false);
    acc2 = __builtin_amdgcn_wmma_f32_16x16x32_f16(false, a, false, b2,
                                                  (short)0, acc2, false, false);
    acc3 = __builtin_amdgcn_wmma_f32_16x16x32_f16(false, a, false, b3,
                                                  (short)0, acc3, false, false);
  }
#pragma unroll
  for (int r = 0; r < 8; ++r) {
    int m = mt * 16 + r + hi * 8;                    // C layout: M = r + 8*hi
    if (m < M) {
      float bv = bias[m];
#pragma unroll
      for (int t = 0; t < 4; ++t) {
        float v = (t == 0 ? acc0[r] : t == 1 ? acc1[r] : t == 2 ? acc2[r] : acc3[r]) + bv;
        if (act == 1) v = v > 0.f ? v : 0.f;
        else if (act == 2) {                         // stable softplus
          float ax = v > 0.f ? v : -v;
          float mx = v > 0.f ? v : 0.f;
          v = mx + log1pf(__expf(-ax));
        }
        size_t o = (size_t)m * NTOT + (nt0 + t) * 16 + lo;
        if (C32) C32[o] = v; else C16[o] = (_Float16)v;
      }
    }
  }
}

// ------------------------------------------- LayerNorm stats (per batch)
__global__ void __launch_bounds__(256)
k_ln_stats(const float* __restrict__ X, int C, long rowStride, long batchStride,
           float* __restrict__ stats) {
  __shared__ double ssum[256], ssq[256];
  const int b = blockIdx.x, t = threadIdx.x;
  const long total = (long)C * PIX;
  double s = 0.0, q = 0.0;
  for (long i = t; i < total; i += 256) {
    int c = (int)(i >> 10), p = (int)(i & 1023);
    float v = X[(long)c * rowStride + (long)b * batchStride + p];
    s += v; q += (double)v * v;
  }
  ssum[t] = s; ssq[t] = q;
  __syncthreads();
  for (int off = 128; off > 0; off >>= 1) {
    if (t < off) { ssum[t] += ssum[t + off]; ssq[t] += ssq[t + off]; }
    __syncthreads();
  }
  if (t == 0) {
    double mu = ssum[0] / (double)total;
    double var = ssq[0] / (double)total - mu * mu;
    if (var < 0.0) var = 0.0;
    stats[b * 2] = (float)mu;
    stats[b * 2 + 1] = (float)(1.0 / sqrt(var + 1e-12));
  }
}

// SPADE apply on x[b][c][p]; gamma/beta f16 plain [c][n]; out B-swizzled f16
__global__ void k_spade_big(const float* __restrict__ X, const _Float16* __restrict__ G,
                            const _Float16* __restrict__ Bt, const float* __restrict__ stats,
                            _Float16* __restrict__ O) {
  int idx = blockIdx.x * 256 + threadIdx.x;   // CIN*NTOT
  if (idx >= CIN * NTOT) return;
  int n = idx % NTOT, c = idx / NTOT;
  int b = n >> 10, p = n & 1023;
  float mu = stats[b * 2], rs = stats[b * 2 + 1];
  float xn = (X[(size_t)b * CIN * PIX + (size_t)c * PIX + p] - mu) * rs;
  O[bswz(c, n)] = (_Float16)(xn * (1.f + (float)G[idx]) + (float)Bt[idx]);
}

// fused: SPADE apply (small C) + 1x1 conv + softplus. H,G,Bt: [Cin][NTOT]
__global__ void k_spade_conv_small(const float* __restrict__ H, const _Float16* __restrict__ G,
                                   const _Float16* __restrict__ Bt, const float* __restrict__ stats,
                                   const float* __restrict__ W, const float* __restrict__ bias,
                                   int Cin, int Cout, float* __restrict__ O) {
  int idx = blockIdx.x * 256 + threadIdx.x;
  if (idx >= Cout * NTOT) return;
  int n = idx % NTOT, oc = idx / NTOT;
  int b = n >> 10;
  float mu = stats[b * 2], rs = stats[b * 2 + 1];
  float acc = bias[oc];
  for (int c = 0; c < Cin; ++c) {
    float hn = (H[c * NTOT + n] - mu) * rs;
    float v = hn * (1.f + (float)G[c * NTOT + n]) + (float)Bt[c * NTOT + n];
    acc += W[oc * Cin + c] * v;
  }
  float ax = acc > 0.f ? acc : -acc;
  float mx = acc > 0.f ? acc : 0.f;
  O[idx] = mx + log1pf(__expf(-ax));
}

// ---------------------------------------------------------------- launcher
extern "C" void kernel_launch(void* const* d_in, const int* in_sizes, int n_in,
                              void* d_out, int out_size, void* d_ws, size_t ws_size,
                              hipStream_t stream) {
  (void)in_sizes; (void)n_in; (void)out_size; (void)ws_size;
  const float* x       = (const float*)d_in[0];
  const int*   segmap  = (const int*)d_in[1];
  const float* s_ws[3] = {(const float*)d_in[2],  (const float*)d_in[8],  (const float*)d_in[14]};
  const float* s_bs[3] = {(const float*)d_in[3],  (const float*)d_in[9],  (const float*)d_in[15]};
  const float* s_wg[3] = {(const float*)d_in[4],  (const float*)d_in[10], (const float*)d_in[16]};
  const float* s_bg[3] = {(const float*)d_in[5],  (const float*)d_in[11], (const float*)d_in[17]};
  const float* s_wb[3] = {(const float*)d_in[6],  (const float*)d_in[12], (const float*)d_in[18]};
  const float* s_bb[3] = {(const float*)d_in[7],  (const float*)d_in[13], (const float*)d_in[19]};
  const float* conv0_w = (const float*)d_in[20];
  const float* bias0   = (const float*)d_in[21];
  const float* conv1_w = (const float*)d_in[22];
  const float* bias1   = (const float*)d_in[23];
  const float* conv2_w = (const float*)d_in[24];
  const float* bias2   = (const float*)d_in[25];

  char* ws = (char*)d_ws;
  size_t off = 0;
  auto take = [&](size_t elems, size_t esz) {
    size_t o = off; off += ((elems * esz + 255) & ~(size_t)255); return o;
  };
  const int nfs[3] = {CIN, 8, 16};
  auto mpad = [](int m) { return (size_t)((m + 15) & ~15); };

  _Float16* wS[3]; for (int i = 0; i < 3; ++i) wS[i] = (_Float16*)(ws + take(mpad(HIDC) * NSEGC * 9, 2));
  _Float16* wG[3]; for (int i = 0; i < 3; ++i) wG[i] = (_Float16*)(ws + take(mpad(nfs[i]) * HIDC * 9, 2));
  _Float16* wB[3]; for (int i = 0; i < 3; ++i) wB[i] = (_Float16*)(ws + take(mpad(nfs[i]) * HIDC * 9, 2));
  _Float16* w0f16  = (_Float16*)(ws + take(mpad(8) * CIN, 2));
  _Float16* segcol = (_Float16*)(ws + take((size_t)9 * NSEGC * NTOT, 2));
  float*    shared = (float*)   (ws + take((size_t)HIDC * NTOT, 4));
  _Float16* shcol  = (_Float16*)(ws + take((size_t)9 * HIDC * NTOT, 2));
  _Float16* gBig   = (_Float16*)(ws + take((size_t)CIN * NTOT, 2));
  _Float16* bBig   = (_Float16*)(ws + take((size_t)CIN * NTOT, 2));
  _Float16* h0col  = (_Float16*)(ws + take((size_t)CIN * NTOT, 2));
  float*    h1     = (float*)   (ws + take((size_t)8 * NTOT, 4));
  _Float16* g1     = (_Float16*)(ws + take((size_t)8 * NTOT, 2));
  _Float16* b1     = (_Float16*)(ws + take((size_t)8 * NTOT, 2));
  float*    h2     = (float*)   (ws + take((size_t)16 * NTOT, 4));
  _Float16* g2     = (_Float16*)(ws + take((size_t)16 * NTOT, 2));
  _Float16* b2     = (_Float16*)(ws + take((size_t)16 * NTOT, 2));
  float*    st0    = (float*)   (ws + take(16, 4));
  float*    st1    = (float*)   (ws + take(16, 4));
  float*    st2    = (float*)   (ws + take(16, 4));

  auto pack = [&](const float* s, _Float16* d, int M, int K) {
    int total = (int)(mpad(M) * K);
    k_pack_A<<<(total + 255) / 256, 256, 0, stream>>>(s, d, M, K);
  };
  for (int i = 0; i < 3; ++i) {
    pack(s_ws[i], wS[i], HIDC, NSEGC * 9);
    pack(s_wg[i], wG[i], nfs[i], HIDC * 9);
    pack(s_wb[i], wB[i], nfs[i], HIDC * 9);
  }
  pack(conv0_w, w0f16, 8, CIN);

  k_segcol<<<(9 * NSEGC * NTOT + 255) / 256, 256, 0, stream>>>(segmap, segcol);
  k_ln_stats<<<NB, 256, 0, stream>>>(x, CIN, PIX, (long)CIN * PIX, st0);

  auto gemm = [&](const _Float16* A, const _Float16* Bm, const float* bias, int M,
                  int K, float* C32, _Float16* C16, int act) {
    dim3 grid(NTOT / 64, (M + 15) / 16);
    k_gemm_wmma<<<grid, dim3(32), 0, stream>>>((const v16h*)A, (const v16h*)Bm,
                                               bias, M, K >> 5, C32, C16, act);
  };

  // ---- SPADE0 (dominant): shared0, im2col, gamma/beta [2048 x 8192]
  gemm(wS[0], segcol, s_bs[0], HIDC, 9 * NSEGC, shared, nullptr, 1);
  k_im2col<<<(9 * HIDC * NTOT + 255) / 256, 256, 0, stream>>>(shared, shcol);
  gemm(wG[0], shcol, s_bg[0], CIN, 9 * HIDC, nullptr, gBig, 0);
  gemm(wB[0], shcol, s_bb[0], CIN, 9 * HIDC, nullptr, bBig, 0);
  k_spade_big<<<(CIN * NTOT + 255) / 256, 256, 0, stream>>>(x, gBig, bBig, st0, h0col);
  gemm(w0f16, h0col, bias0, 8, CIN, h1, nullptr, 2);   // conv0 1x1 + softplus

  // ---- SPADE1
  k_ln_stats<<<NB, 256, 0, stream>>>(h1, 8, NTOT, PIX, st1);
  gemm(wS[1], segcol, s_bs[1], HIDC, 9 * NSEGC, shared, nullptr, 1);
  k_im2col<<<(9 * HIDC * NTOT + 255) / 256, 256, 0, stream>>>(shared, shcol);
  gemm(wG[1], shcol, s_bg[1], 8, 9 * HIDC, nullptr, g1, 0);
  gemm(wB[1], shcol, s_bb[1], 8, 9 * HIDC, nullptr, b1, 0);
  k_spade_conv_small<<<(16 * NTOT + 255) / 256, 256, 0, stream>>>(
      h1, g1, b1, st1, conv1_w, bias1, 8, 16, h2);

  // ---- SPADE2
  k_ln_stats<<<NB, 256, 0, stream>>>(h2, 16, NTOT, PIX, st2);
  gemm(wS[2], segcol, s_bs[2], HIDC, 9 * NSEGC, shared, nullptr, 1);
  k_im2col<<<(9 * HIDC * NTOT + 255) / 256, 256, 0, stream>>>(shared, shcol);
  gemm(wG[2], shcol, s_bg[2], 16, 9 * HIDC, nullptr, g2, 0);
  gemm(wB[2], shcol, s_bb[2], 16, 9 * HIDC, nullptr, b2, 0);
  k_spade_conv_small<<<(1 * NTOT + 255) / 256, 256, 0, stream>>>(
      h2, g2, b2, st2, conv2_w, bias2, 16, 1, (float*)d_out);
}
